// BBoxHeadForGroundTruthBboxRegressionV1_25228637897254
// MI455X (gfx1250) — matrix operand order repaired and checked by reading, not compiled, not run on hardware
//
#include <hip/hip_runtime.h>
#include <hip/hip_bf16.h>
#include <math.h>

// Problem constants (match reference)
#define B_    8
#define L_    256
#define T_    2048
#define LM_   4096
#define DFF_  1024
#define D_    84
#define H_    4
#define HD_   21
#define D3_   (3*D_)      // 252
#define VISF  (L_*D_)     // 21504
#define FIN_  (VISF+LM_)  // 25600

typedef __attribute__((ext_vector_type(16))) __bf16        v16bf;
typedef __attribute__((ext_vector_type(8)))  float         v8f;
typedef __attribute__((ext_vector_type(4)))  unsigned int  v4u;
typedef __attribute__((ext_vector_type(8)))  int           v8i;
typedef __attribute__((ext_vector_type(4)))  int           v4i;

// ---------------------------------------------------------------------------
// Kernel 1: fourier positional embedding + class-embedding gather -> src[B*L,84]
// ---------------------------------------------------------------------------
__global__ void k_build_src(const float* __restrict__ vf, const float* __restrict__ gB,
                            const float* __restrict__ cemb, float* __restrict__ src,
                            float* __restrict__ neg)
{
  int i = blockIdx.x * blockDim.x + threadIdx.x;
  if (i >= B_ * L_) return;
  const float* v = vf + i * 7;
  float x0 = v[0], x1 = v[1], x2 = v[2];
  float y0 = v[3], y1 = v[4], y2 = v[5];
  float c  = v[6];
  bool pad = (x0==0.f)&&(x1==0.f)&&(x2==0.f)&&(y0==0.f)&&(y1==0.f)&&(y2==0.f)&&(c==0.f);
  neg[i] = pad ? -1e9f : 0.f;
  float* s = src + i * D_;
  const float TWO_PI = 6.283185307179586f;
#pragma unroll
  for (int d = 0; d < 5; ++d) {
    float p = TWO_PI * (x0 * gB[0*5+d] + x1 * gB[1*5+d] + x2 * gB[2*5+d]);
    s[d]     = sinf(p);
    s[5 + d] = cosf(p);
  }
#pragma unroll
  for (int d = 0; d < 5; ++d) {
    float p = TWO_PI * (y0 * gB[0*5+d] + y1 * gB[1*5+d] + y2 * gB[2*5+d]);
    s[10 + d] = sinf(p);
    s[15 + d] = cosf(p);
  }
  int cls = (int)c;
  const float* e = cemb + (size_t)cls * 64;
#pragma unroll 8
  for (int j = 0; j < 64; ++j) s[20 + j] = e[j];
}

// ---------------------------------------------------------------------------
// Encoder: qkv projection [2048,84]@[84,252]
// ---------------------------------------------------------------------------
__global__ void k_qkv(const float* __restrict__ src, const float* __restrict__ w,
                      const float* __restrict__ b, float* __restrict__ qkv)
{
  int i = blockIdx.x * blockDim.x + threadIdx.x;
  if (i >= B_ * L_ * D3_) return;
  int row = i / D3_, j = i % D3_;
  const float* s = src + row * D_;
  float acc = b[j];
#pragma unroll 4
  for (int k = 0; k < D_; ++k) acc += s[k] * w[k * D3_ + j];
  qkv[i] = acc;
}

// ---------------------------------------------------------------------------
// Encoder: fused attention (one thread per (b,h,l), 3-pass softmax)
// ---------------------------------------------------------------------------
__global__ void k_attn(const float* __restrict__ qkv, const float* __restrict__ neg,
                       float* __restrict__ o)
{
  int i = blockIdx.x * blockDim.x + threadIdx.x;
  if (i >= B_ * H_ * L_) return;
  int b = i / (H_ * L_);
  int h = (i / L_) % H_;
  int l = i % L_;
  const float scale = rsqrtf((float)HD_);
  const float* qp = qkv + ((size_t)(b*L_+l) * D3_) + h * HD_;
  float q[HD_];
#pragma unroll
  for (int d = 0; d < HD_; ++d) q[d] = qp[d] * scale;
  const float* kbase = qkv + (size_t)b * L_ * D3_ + D_   + h * HD_;
  const float* vbase = qkv + (size_t)b * L_ * D3_ + 2*D_ + h * HD_;
  const float* nb = neg + b * L_;
  float mx = -1e30f;
  for (int m = 0; m < L_; ++m) {
    const float* kp = kbase + (size_t)m * D3_;
    float sc = nb[m];
#pragma unroll
    for (int d = 0; d < HD_; ++d) sc += q[d] * kp[d];
    mx = fmaxf(mx, sc);
  }
  float sum = 0.f;
  for (int m = 0; m < L_; ++m) {
    const float* kp = kbase + (size_t)m * D3_;
    float sc = nb[m];
#pragma unroll
    for (int d = 0; d < HD_; ++d) sc += q[d] * kp[d];
    sum += expf(sc - mx);
  }
  float inv = 1.f / sum;
  float ov[HD_];
#pragma unroll
  for (int d = 0; d < HD_; ++d) ov[d] = 0.f;
  for (int m = 0; m < L_; ++m) {
    const float* kp = kbase + (size_t)m * D3_;
    float sc = nb[m];
#pragma unroll
    for (int d = 0; d < HD_; ++d) sc += q[d] * kp[d];
    float w = expf(sc - mx) * inv;
    const float* vp = vbase + (size_t)m * D3_;
#pragma unroll
    for (int d = 0; d < HD_; ++d) ov[d] += w * vp[d];
  }
  float* op = o + (size_t)(b*L_+l) * D_ + h * HD_;
#pragma unroll
  for (int d = 0; d < HD_; ++d) op[d] = ov[d];
}

// ---------------------------------------------------------------------------
// Encoder: out-proj + residual + LayerNorm (one block per row, 128 threads)
// ---------------------------------------------------------------------------
__global__ void k_projln(const float* __restrict__ o, const float* __restrict__ w,
                         const float* __restrict__ bo, const float* __restrict__ res,
                         const float* __restrict__ gs, const float* __restrict__ gb,
                         float* __restrict__ out)
{
  int row = blockIdx.x;
  int j = threadIdx.x;
  __shared__ float sh[128];
  float val = 0.f;
  if (j < D_) {
    const float* op = o + (size_t)row * D_;
    float acc = bo[j] + res[(size_t)row * D_ + j];
#pragma unroll 4
    for (int k = 0; k < D_; ++k) acc += op[k] * w[k * D_ + j];
    val = acc;
  }
  sh[j] = (j < D_) ? val : 0.f;
  __syncthreads();
  for (int st = 64; st > 0; st >>= 1) { if (j < st) sh[j] += sh[j + st]; __syncthreads(); }
  float mean = sh[0] / (float)D_;
  __syncthreads();
  float dv = val - mean;
  sh[j] = (j < D_) ? dv * dv : 0.f;
  __syncthreads();
  for (int st = 64; st > 0; st >>= 1) { if (j < st) sh[j] += sh[j + st]; __syncthreads(); }
  float var = sh[0] / (float)D_;
  if (j < D_)
    out[(size_t)row * D_ + j] = dv * rsqrtf(var + 1e-5f) * gs[j] + gb[j];
}

// ---------------------------------------------------------------------------
// Encoder: FF1 + relu [2048,84]@[84,1024]
// ---------------------------------------------------------------------------
__global__ void k_ff1(const float* __restrict__ src, const float* __restrict__ w,
                      const float* __restrict__ b, float* __restrict__ out)
{
  int i = blockIdx.x * blockDim.x + threadIdx.x;
  if (i >= B_ * L_ * DFF_) return;
  int row = i / DFF_, j = i % DFF_;
  const float* s = src + (size_t)row * D_;
  float acc = b[j];
#pragma unroll 4
  for (int k = 0; k < D_; ++k) acc += s[k] * w[k * DFF_ + j];
  out[i] = fmaxf(acc, 0.f);
}

// ---------------------------------------------------------------------------
// Encoder: FF2 + residual + LayerNorm (block per row)
// ---------------------------------------------------------------------------
__global__ void k_ff2ln(const float* __restrict__ ff, const float* __restrict__ w,
                        const float* __restrict__ b2, const float* __restrict__ res,
                        const float* __restrict__ gs, const float* __restrict__ gb,
                        float* __restrict__ out)
{
  int row = blockIdx.x;
  int j = threadIdx.x;
  __shared__ float sh[128];
  float val = 0.f;
  if (j < D_) {
    const float* fp = ff + (size_t)row * DFF_;
    float acc = b2[j] + res[(size_t)row * D_ + j];
    for (int k = 0; k < DFF_; ++k) acc += fp[k] * w[k * D_ + j];
    val = acc;
  }
  sh[j] = (j < D_) ? val : 0.f;
  __syncthreads();
  for (int st = 64; st > 0; st >>= 1) { if (j < st) sh[j] += sh[j + st]; __syncthreads(); }
  float mean = sh[0] / (float)D_;
  __syncthreads();
  float dv = val - mean;
  sh[j] = (j < D_) ? dv * dv : 0.f;
  __syncthreads();
  for (int st = 64; st > 0; st >>= 1) { if (j < st) sh[j] += sh[j + st]; __syncthreads(); }
  float var = sh[0] / (float)D_;
  if (j < D_)
    out[(size_t)row * D_ + j] = dv * rsqrtf(var + 1e-5f) * gs[j] + gb[j];
}

// ---------------------------------------------------------------------------
// Pack vision_flat rows into bf16 A matrix padded to M=16 (rows 8..15 zero)
// ---------------------------------------------------------------------------
__global__ void k_pack_vis(const float* __restrict__ src, __bf16* __restrict__ Apad)
{
  int i = blockIdx.x * blockDim.x + threadIdx.x;
  if (i >= 16 * VISF) return;
  int r = i / VISF, c = i % VISF;
  Apad[i] = (r < B_) ? (__bf16)src[(size_t)r * VISF + c] : (__bf16)0.f;
}

__global__ void k_cvt_bf16(const float* __restrict__ x, __bf16* __restrict__ y, int n)
{
  int i = blockIdx.x * blockDim.x + threadIdx.x;
  if (i < n) y[i] = (__bf16)x[i];
}

// ---------------------------------------------------------------------------
// Generic bf16 WMMA GEMM: C[M,N] = act(A[M,K](bf16) @ B[K,N](f32->bf16) + bias
//                                     + addRows[gatherIdx[m]])
// - A tile (64x32 bf16) fetched by the Tensor Data Mover (tensor_load_to_lds);
//   tensor_dim OOB clamping gives free zero-padding when M < 64.
// - B tile converted f32->bf16 and stored TRANSPOSED in LDS so both WMMA
//   operand fragments read k as two 16B runs (ds_load_b128 pairs).
// block tile 64x128, 8 waves (2x4), each wave 32x32 via 2x2 v_wmma 16x16x32 frags
// ---------------------------------------------------------------------------
__global__ __launch_bounds__(256)
void k_wmma_gemm(const __bf16* __restrict__ A, const float* __restrict__ Bw,
                 const float* __restrict__ bias, const float* __restrict__ addRows,
                 const int* __restrict__ gatherIdx, float* __restrict__ Cf,
                 __bf16* __restrict__ Cb, int M, int N, int K, int relu)
{
  constexpr int BM = 64, BN = 128, BK = 32, BKP = BK + 8; // padded stride (80B, 16B-aligned)
  __shared__ __bf16 As[BM][BK];     // 4 KB, filled by TDM
  __shared__ __bf16 BsT[BN][BKP];   // 10 KB, transposed B tile
  const int tid  = threadIdx.x;
  const int lane = tid & 31;
  const int wave = tid >> 5;
  const int wm = wave >> 2;       // 0..1  (M direction)
  const int wn = wave & 3;        // 0..3  (N direction)
  const int grp = lane >> 4;      // half-wave group
  const int lr  = lane & 15;
  const int blockM = blockIdx.y * BM;
  const int blockN = blockIdx.x * BN;

  v8f cacc[2][2];
  const v8f vz = {0.f,0.f,0.f,0.f,0.f,0.f,0.f,0.f};
#pragma unroll
  for (int s = 0; s < 2; ++s)
#pragma unroll
    for (int t = 0; t < 2; ++t) cacc[s][t] = vz;

  const unsigned int lds_a = (unsigned int)(unsigned long long)(const void*)&As[0][0];
  const int brow = tid >> 3;             // 0..31 (B stage row)
  const int bcol = (tid & 7) * 16;       // 0..112

  for (int k0 = 0; k0 < K; k0 += BK) {
    // ---- issue TDM load of the A tile (wave 0 only; EXEC ignored by TDM) ----
    if (wave == 0) {
      unsigned long long ga =
          (unsigned long long)(const void*)(A + (size_t)blockM * K + k0);
      v4u g0;
      g0.x = 1u;                                   // count=1 (valid user D#)
      g0.y = lds_a;                                // lds_addr
      g0.z = (unsigned int)ga;                     // global_addr[31:0]
      g0.w = (unsigned int)(ga >> 32) | (2u << 30);// global_addr[56:32] | type=2
      int td0 = K - k0;                            // remaining X extent
      int td1 = M - blockM;                        // remaining Y extent (OOB rows -> 0)
      v8i g1;
      g1[0] = (1 << 16);                                          // data_size=2B
      g1[1] = (td0 & 0xFFFF) << 16;                               // tensor_dim0 lo
      g1[2] = ((td0 >> 16) & 0xFFFF) | ((td1 & 0xFFFF) << 16);    // dim0 hi | dim1 lo
      g1[3] = ((td1 >> 16) & 0xFFFF) | (BK << 16);                // dim1 hi | tile_dim0
      g1[4] = BM;                                                 // tile_dim1 | tile_dim2=0
      g1[5] = K;                                                  // tensor_dim0_stride lo
      g1[6] = 0;                                                  // stride0 hi | stride1 lo
      g1[7] = 0;
      v4i zz4 = {0, 0, 0, 0};
      v8i zz8 = {0, 0, 0, 0, 0, 0, 0, 0};
      __builtin_amdgcn_tensor_load_to_lds(g0, g1, zz4, zz4, zz8, 0);
    }
    // ---- stage B tile transposed (f32 -> bf16, 16 elems per thread) ----
    {
      const float* bp = Bw + (size_t)(k0 + brow) * N + blockN + bcol;
#pragma unroll
      for (int q = 0; q < 16; ++q) BsT[bcol + q][brow] = (__bf16)bp[q];
      if (k0 + BK < K)   // prefetch next B tile (global_prefetch_b8)
        __builtin_prefetch(Bw + (size_t)(k0 + BK + brow) * N + blockN + bcol, 0, 0);
    }
    if (wave == 0) __builtin_amdgcn_s_wait_tensorcnt(0);
    __syncthreads();

    // ---- build fragments (both operands two 16B k-runs per lane) ----
    v16bf af[2], bfr[2];
#pragma unroll
    for (int s = 0; s < 2; ++s) {
      int row = wm * 32 + s * 16 + lr;
#pragma unroll
      for (int e = 0; e < 16; ++e) {
        int kk = e + 8 * ((e >> 3) + grp);   // A 16x32 bf16 layout (ISA 7.12.2)
        af[s][e] = As[row][kk];
      }
    }
#pragma unroll
    for (int t = 0; t < 2; ++t) {
      int col = wn * 32 + t * 16 + lr;
#pragma unroll
      for (int e = 0; e < 16; ++e) {
        int kk = e + 8 * ((e >> 3) + grp);   // B 32x16 bf16 layout
        bfr[t][e] = BsT[col][kk];
      }
    }

#pragma unroll
    for (int s = 0; s < 2; ++s)
#pragma unroll
      for (int t = 0; t < 2; ++t)
        cacc[s][t] = __builtin_amdgcn_wmma_f32_16x16x32_bf16(
            false, af[s], false, bfr[t], (short)0, cacc[s][t], false, false);
    __syncthreads();
  }

  // ---- epilogue: bias + gathered add + relu, store f32 and/or bf16 ----
#pragma unroll
  for (int s = 0; s < 2; ++s) {
#pragma unroll
    for (int t = 0; t < 2; ++t) {
      int gn = blockN + wn * 32 + t * 16 + lr;
#pragma unroll
      for (int v = 0; v < 8; ++v) {
        int gm = blockM + wm * 32 + s * 16 + grp * 8 + v;  // C/D: VGPR v -> M = v + 8*grp
        if (gm < M) {
          float x = cacc[s][t][v];
          if (bias)    x += bias[gn];
          if (addRows) x += addRows[(size_t)gatherIdx[gm] * N + gn];
          if (relu)    x = fmaxf(x, 0.f);
          if (Cf) Cf[(size_t)gm * N + gn] = x;
          if (Cb) Cb[(size_t)gm * N + gn] = (__bf16)x;
        }
      }
    }
  }
}

// ---------------------------------------------------------------------------
// Final head: out[2048,6] = h4(bf16) @ w4 + b4
// ---------------------------------------------------------------------------
__global__ void k_head_out(const __bf16* __restrict__ h, const float* __restrict__ w4,
                           const float* __restrict__ b4, float* __restrict__ out)
{
  int i = blockIdx.x * blockDim.x + threadIdx.x;
  if (i >= T_ * 6) return;
  int row = i / 6, j = i % 6;
  const __bf16* hp = h + (size_t)row * DFF_;
  float acc = b4[j];
#pragma unroll 4
  for (int k = 0; k < DFF_; ++k) acc += (float)hp[k] * w4[k * 6 + j];
  out[i] = acc;
}

// ---------------------------------------------------------------------------
extern "C" void kernel_launch(void* const* d_in, const int* in_sizes, int n_in,
                              void* d_out, int out_size, void* d_ws, size_t ws_size,
                              hipStream_t stream)
{
  (void)in_sizes; (void)n_in; (void)out_size; (void)ws_size;
  const float* grd   = (const float*)d_in[0];
  const float* vf    = (const float*)d_in[1];
  const int*   tbi   = (const int*)  d_in[2];
  const float* gB    = (const float*)d_in[3];
  const float* cemb  = (const float*)d_in[4];
  const float* w_in  = (const float*)d_in[5];
  const float* b_in  = (const float*)d_in[6];
  const float* w_out = (const float*)d_in[7];
  const float* b_out = (const float*)d_in[8];
  const float* ln1s  = (const float*)d_in[9];
  const float* ln1b  = (const float*)d_in[10];
  const float* wff1  = (const float*)d_in[11];
  const float* bff1  = (const float*)d_in[12];
  const float* wff2  = (const float*)d_in[13];
  const float* bff2  = (const float*)d_in[14];
  const float* ln2s  = (const float*)d_in[15];
  const float* ln2b  = (const float*)d_in[16];
  const float* w0    = (const float*)d_in[17];
  const float* b0    = (const float*)d_in[18];
  const float* w1    = (const float*)d_in[19];
  const float* b1    = (const float*)d_in[20];
  const float* w2    = (const float*)d_in[21];
  const float* b2    = (const float*)d_in[22];
  const float* w3    = (const float*)d_in[23];
  const float* b3    = (const float*)d_in[24];
  const float* w4    = (const float*)d_in[25];
  const float* b4    = (const float*)d_in[26];
  float* out = (float*)d_out;

  // workspace bump allocator (256B aligned)
  size_t off = 0;
  char* base = (char*)d_ws;
  auto alloc = [&](size_t bytes) -> char* {
    char* p = base + off;
    off += (bytes + 255) & ~(size_t)255;
    return p;
  };
  float*  src  = (float*) alloc((size_t)T_ * D_ * 4);
  float*  src2 = (float*) alloc((size_t)T_ * D_ * 4);
  float*  neg  = (float*) alloc((size_t)B_ * L_ * 4);
  float*  qkv  = (float*) alloc((size_t)B_ * L_ * D3_ * 4);
  float*  atto = (float*) alloc((size_t)B_ * L_ * D_ * 4);
  float*  ffb  = (float*) alloc((size_t)B_ * L_ * DFF_ * 4);
  __bf16* Apad = (__bf16*)alloc((size_t)16 * VISF * 2);
  float*  tmp8 = (float*) alloc((size_t)16 * DFF_ * 4);
  __bf16* grdb = (__bf16*)alloc((size_t)T_ * LM_ * 2);
  __bf16* h1   = (__bf16*)alloc((size_t)T_ * DFF_ * 2);
  __bf16* h2   = (__bf16*)alloc((size_t)T_ * DFF_ * 2);

  // --- build src + padding bias ---
  k_build_src<<<(B_*L_ + 255)/256, 256, 0, stream>>>(vf, gB, cemb, src, neg);

  // --- 2-layer post-norm encoder (fp32 VALU; ~2% of total FLOPs) ---
  for (int l = 0; l < 2; ++l) {
    k_qkv<<<(B_*L_*D3_ + 255)/256, 256, 0, stream>>>(
        src, w_in + (size_t)l*D_*D3_, b_in + (size_t)l*D3_, qkv);
    k_attn<<<(B_*H_*L_ + 255)/256, 256, 0, stream>>>(qkv, neg, atto);
    k_projln<<<B_*L_, 128, 0, stream>>>(
        atto, w_out + (size_t)l*D_*D_, b_out + (size_t)l*D_, src,
        ln1s + (size_t)l*D_, ln1b + (size_t)l*D_, src2);
    k_ff1<<<(B_*L_*DFF_ + 255)/256, 256, 0, stream>>>(
        src2, wff1 + (size_t)l*D_*DFF_, bff1 + (size_t)l*DFF_, ffb);
    k_ff2ln<<<B_*L_, 128, 0, stream>>>(
        ffb, wff2 + (size_t)l*DFF_*D_, bff2 + (size_t)l*D_, src2,
        ln2s + (size_t)l*D_, ln2b + (size_t)l*D_, src);
  }

  // --- pack vision_flat (8 rows, padded to 16) and grd_tokens to bf16 ---
  k_pack_vis<<<(16*VISF + 255)/256, 256, 0, stream>>>(src, Apad);
  k_cvt_bf16<<<((T_*LM_) + 255)/256, 256, 0, stream>>>(grd, grdb, T_*LM_);

  // --- GEMM1: tmp8[16,1024] = Apad[16,21504] @ w0[:21504,:]  (no bias) ---
  {
    dim3 g(DFF_/128, (16 + 63)/64);
    k_wmma_gemm<<<g, 256, 0, stream>>>(Apad, w0, nullptr, nullptr, nullptr,
                                       tmp8, nullptr, 16, DFF_, VISF, 0);
  }
  // --- GEMM2: h1 = relu(grd @ w0[21504:,:] + tmp8[bidx] + b0) ---
  {
    dim3 g(DFF_/128, T_/64);
    k_wmma_gemm<<<g, 256, 0, stream>>>(grdb, w0 + (size_t)VISF*DFF_, b0, tmp8, tbi,
                                       nullptr, h1, T_, DFF_, LM_, 1);
  }
  // --- GEMM3..5: DFF x DFF chain ---
  {
    dim3 g(DFF_/128, T_/64);
    k_wmma_gemm<<<g, 256, 0, stream>>>(h1, w1, b1, nullptr, nullptr,
                                       nullptr, h2, T_, DFF_, DFF_, 1);
    k_wmma_gemm<<<g, 256, 0, stream>>>(h2, w2, b2, nullptr, nullptr,
                                       nullptr, h1, T_, DFF_, DFF_, 1);
    k_wmma_gemm<<<g, 256, 0, stream>>>(h1, w3, b3, nullptr, nullptr,
                                       nullptr, h2, T_, DFF_, DFF_, 1);
  }
  // --- final head: [2048,1024] @ [1024,6] + b4 ---
  k_head_out<<<(T_*6 + 255)/256, 256, 0, stream>>>(h2, w4, b4, out);
}
